// SparseChangeTransformer_27255862460631
// MI455X (gfx1250) — matrix operand order repaired and checked by reading, not compiled, not run on hardware
//
#include <hip/hip_runtime.h>
#include <hip/hip_bf16.h>
#include <math.h>

typedef __attribute__((ext_vector_type(16))) __bf16 v16bf;
typedef __attribute__((ext_vector_type(8)))  __bf16 v8bf;
typedef __attribute__((ext_vector_type(8)))  float  v8f;

#define B_   4
#define C_   192
#define H_   128
#define W_   128
#define N_   (H_*W_)     // 16384
#define KSEL 2048
#define HEADS 3
#define HD   64
#define HID_ 768
#define EPS_ 1e-5f

// ---------------------------------------------------------------- DPP reductions
// 16-lane reduction network (matches WMMA C-layout half-groups):
//   quad_perm[1,0,3,2]=0xB1, quad_perm[2,3,0,1]=0x4E, ROW_HALF_MIRROR=0x141, ROW_MIRROR=0x140
template <int CTRL>
static __device__ __forceinline__ float dpp_movf(float v) {
    return __int_as_float(
        __builtin_amdgcn_update_dpp(0, __float_as_int(v), CTRL, 0xF, 0xF, true));
}
static __device__ __forceinline__ float red16_max(float v) {
    v = fmaxf(v, dpp_movf<0xB1>(v));
    v = fmaxf(v, dpp_movf<0x4E>(v));
    v = fmaxf(v, dpp_movf<0x141>(v));
    v = fmaxf(v, dpp_movf<0x140>(v));
    return v;
}
static __device__ __forceinline__ float red16_sum(float v) {
    v += dpp_movf<0xB1>(v);
    v += dpp_movf<0x4E>(v);
    v += dpp_movf<0x141>(v);
    v += dpp_movf<0x140>(v);
    return v;
}
static __device__ __forceinline__ float red32_sum(float v) {
    v = red16_sum(v);
    int i = __float_as_int(v);
    int x = __builtin_amdgcn_permlanex16(i, i, 0x76543210, 0xFEDCBA98, false, false);
    return v + __int_as_float(x);
}

// ---------------------------------------------------------------- async copy helpers
// GLOBAL_LOAD_ASYNC_TO_LDS_B128 (GVS mode): 16 bytes per lane, global -> LDS,
// tracked by ASYNCcnt. LDS address = low 32 bits of the generic pointer.
static __device__ __forceinline__ void async_g2l_16B(const __bf16* gbase, int byte_off,
                                                     const __bf16* lds_ptr) {
    unsigned lds = (unsigned)(size_t)lds_ptr;
    asm volatile("global_load_async_to_lds_b128 %0, %1, %2"
                 :: "v"(lds), "v"(byte_off), "s"(gbase)
                 : "memory");
}
static __device__ __forceinline__ void wait_async0() {
    asm volatile("s_wait_asynccnt 0x0" ::: "memory");
}

// ---------------------------------------------------------------- utilities
__global__ void f32_to_bf16_kernel(const float* __restrict__ in,
                                   __bf16* __restrict__ out, int n) {
    int i = blockIdx.x * blockDim.x + threadIdx.x;
    if (i < n) out[i] = (__bf16)in[i];
}

// LN1: x (B,C,H,W) -> y (B,N,C) fp32, one wave per token
__global__ __launch_bounds__(256) void ln1_kernel(
    const float* __restrict__ x, const float* __restrict__ w,
    const float* __restrict__ b, float* __restrict__ y) {
    int gw   = (blockIdx.x * blockDim.x + threadIdx.x) >> 5;   // token id over B*N
    int lane = threadIdx.x & 31;
    int bi = gw / N_, n = gw % N_;
    const float* xb = x + (size_t)bi * C_ * N_ + n;
    float v[6], s = 0.f;
    #pragma unroll
    for (int i = 0; i < 6; ++i) { int c = lane + 32 * i; v[i] = xb[(size_t)c * N_]; s += v[i]; }
    float mu = red32_sum(s) * (1.f / C_);
    float vs = 0.f;
    #pragma unroll
    for (int i = 0; i < 6; ++i) { float d = v[i] - mu; vs += d * d; }
    float inv = rsqrtf(red32_sum(vs) * (1.f / C_) + EPS_);
    float* yr = y + (size_t)gw * C_;
    #pragma unroll
    for (int i = 0; i < 6; ++i) { int c = lane + 32 * i; yr[c] = (v[i] - mu) * inv * w[c] + b[c]; }
}

// gather selected rows of y -> bf16 (B,K,C)
__global__ void gather_kernel(const float* __restrict__ y,
                              const int* __restrict__ idx,
                              __bf16* __restrict__ sel) {
    int bk = blockIdx.x;            // b*KSEL + k
    int bi = bk / KSEL;
    int t  = idx[bk];
    sel[(size_t)bk * C_ + threadIdx.x] =
        (__bf16)y[((size_t)bi * N_ + t) * C_ + threadIdx.x];
}

// tokens = x + y (y has attn rows scattered in); t2 = LN2(tokens); fp32 + bf16
__global__ __launch_bounds__(256) void resid_ln2_kernel(
    const float* __restrict__ x, const float* __restrict__ y,
    const float* __restrict__ w, const float* __restrict__ b,
    float* __restrict__ t2f, __bf16* __restrict__ t2b) {
    int gw   = (blockIdx.x * blockDim.x + threadIdx.x) >> 5;
    int lane = threadIdx.x & 31;
    int bi = gw / N_, n = gw % N_;
    const float* xb = x + (size_t)bi * C_ * N_ + n;
    const float* yr = y + (size_t)gw * C_;
    float v[6], s = 0.f;
    #pragma unroll
    for (int i = 0; i < 6; ++i) {
        int c = lane + 32 * i;
        v[i] = xb[(size_t)c * N_] + yr[c];
        s += v[i];
    }
    float mu = red32_sum(s) * (1.f / C_);
    float vs = 0.f;
    #pragma unroll
    for (int i = 0; i < 6; ++i) { float d = v[i] - mu; vs += d * d; }
    float inv = rsqrtf(red32_sum(vs) * (1.f / C_) + EPS_);
    #pragma unroll
    for (int i = 0; i < 6; ++i) {
        int c = lane + 32 * i;
        float o = (v[i] - mu) * inv * w[c] + b[c];
        t2f[(size_t)gw * C_ + c] = o;
        t2b[(size_t)gw * C_ + c] = (__bf16)o;
    }
}

// depthwise 3x3 conv (SAME) + exact GELU; h1 (B,HID,H,W) bf16 -> h1g (B,N,HID) bf16
__global__ __launch_bounds__(256) void dwconv_gelu_kernel(
    const __bf16* __restrict__ h1, const float* __restrict__ wdw,
    __bf16* __restrict__ h1g) {
    size_t gid = (size_t)blockIdx.x * blockDim.x + threadIdx.x;   // ((b*HID+ch)*N + n)
    int n  = (int)(gid % N_);
    int ch = (int)((gid / N_) % HID_);
    int bi = (int)(gid / ((size_t)HID_ * N_));
    int xw = n % W_, yh = n / W_;
    const __bf16* plane = h1 + (size_t)(bi * HID_ + ch) * N_;
    const float* wc = wdw + ch * 9;
    float acc = 0.f;
    #pragma unroll
    for (int dy = 0; dy < 3; ++dy) {
        int yy = yh + dy - 1;
        if (yy < 0 || yy >= H_) continue;
        #pragma unroll
        for (int dx = 0; dx < 3; ++dx) {
            int xx = xw + dx - 1;
            if (xx < 0 || xx >= W_) continue;
            acc += (float)plane[yy * W_ + xx] * wc[dy * 3 + dx];
        }
    }
    float g = 0.5f * acc * (1.f + erff(acc * 0.70710678118654752f));
    h1g[((size_t)bi * N_ + n) * HID_ + ch] = (__bf16)g;
}

// ---------------------------------------------------------------- generic WMMA GEMM
// out = A(MxKd, bf16 row-major) @ W^T (W is Nd x Kd bf16 row-major), fp32 accum.
// Block tile 128x64, 8 waves of 32x32. Double-buffered async global->LDS staging.
// MODE 0: qkv  -> store bf16 into (B,9,K,64) planes
// MODE 1: proj -> +bias, scatter fp32 rows into y via indices
// MODE 2: fc1  -> +bias, store bf16 transposed (B,HID,N)
// MODE 3: fc2  -> +bias +resid(t2), store fp32 final (B,C,N)
template <int MODE>
__global__ __launch_bounds__(256) void gemm_kernel(
    const __bf16* __restrict__ Abase, const __bf16* __restrict__ Wm,
    const float* __restrict__ bias, int M, int Kd,
    void* __restrict__ outp, const int* __restrict__ idx,
    const float* __restrict__ resid) {
    __shared__ __bf16 As[2][128][32];
    __shared__ __bf16 Ws[2][64][32];
    int tid = threadIdx.x, lane = tid & 31, wave = tid >> 5;
    int half = lane >> 4, mlane = lane & 15;
    int z = blockIdx.z;
    int row0 = blockIdx.x * 128, col0 = blockIdx.y * 64;
    const __bf16* A = Abase + (size_t)z * M * Kd;
    int wrow = wave >> 1, wcol = wave & 1;

    // async staging of one K-tile (A: 2 x 16B per thread, W: 1 x 16B per thread)
    auto issue_tile = [&](int buf, int kt) {
        #pragma unroll
        for (int j = 0; j < 2; ++j) {
            int s = tid + j * 256;          // 512 slots: row = s/4, 8-elem part = s%4
            int r = s >> 2, part = (s & 3) * 8;
            async_g2l_16B(A, ((row0 + r) * Kd + kt + part) * 2, &As[buf][r][part]);
        }
        {
            int r = tid >> 2, part = (tid & 3) * 8;   // 256 slots: 64 rows x 4 parts
            async_g2l_16B(Wm, ((col0 + r) * Kd + kt + part) * 2, &Ws[buf][r][part]);
        }
    };

    v8f c[2][2];
    #pragma unroll
    for (int i = 0; i < 2; ++i)
        #pragma unroll
        for (int j = 0; j < 2; ++j)
            #pragma unroll
            for (int r = 0; r < 8; ++r) c[i][j][r] = 0.f;

    int ntile = Kd / 32;
    issue_tile(0, 0);
    wait_async0();
    __syncthreads();

    for (int t = 0; t < ntile; ++t) {
        int cur = t & 1;
        if (t + 1 < ntile) issue_tile(cur ^ 1, (t + 1) * 32);

        v16bf a[2], bf[2];
        #pragma unroll
        for (int mi = 0; mi < 2; ++mi) {
            const __bf16* base = &As[cur][wrow * 32 + mi * 16 + mlane][0];
            v8bf lo = *(const v8bf*)(base + half * 16);
            v8bf hi = *(const v8bf*)(base + half * 16 + 8);
            #pragma unroll
            for (int j = 0; j < 8; ++j) { a[mi][j] = lo[j]; a[mi][8 + j] = hi[j]; }
        }
        #pragma unroll
        for (int ni = 0; ni < 2; ++ni)
            bf[ni] = *(const v16bf*)&Ws[cur][wcol * 32 + ni * 16 + mlane][half * 16];

        #pragma unroll
        for (int mi = 0; mi < 2; ++mi)
            #pragma unroll
            for (int ni = 0; ni < 2; ++ni)
                c[mi][ni] = __builtin_amdgcn_wmma_f32_16x16x32_bf16(
                    false, a[mi], false, bf[ni], (short)0, c[mi][ni], false, false);

        wait_async0();       // next tile landed (own loads; barrier covers the rest)
        __syncthreads();
    }

    #pragma unroll
    for (int mi = 0; mi < 2; ++mi)
        #pragma unroll
        for (int ni = 0; ni < 2; ++ni)
            #pragma unroll
            for (int r = 0; r < 8; ++r) {
                int grow = row0 + wrow * 32 + mi * 16 + r + half * 8;
                int gcol = col0 + wcol * 32 + ni * 16 + mlane;
                float val = c[mi][ni][r];
                if (MODE == 0) {
                    ((__bf16*)outp)[((size_t)(z * 9 + (gcol >> 6)) * KSEL + grow) * 64 + (gcol & 63)] =
                        (__bf16)val;
                } else if (MODE == 1) {
                    val += bias[gcol];
                    int t2 = idx[z * KSEL + grow];
                    ((float*)outp)[((size_t)z * N_ + t2) * C_ + gcol] = val;
                } else if (MODE == 2) {
                    ((__bf16*)outp)[((size_t)(z * HID_ + gcol)) * N_ + grow] =
                        (__bf16)(val + bias[gcol]);
                } else {
                    val += bias[gcol] + resid[((size_t)z * N_ + grow) * C_ + gcol];
                    ((float*)outp)[((size_t)(z * C_ + gcol)) * N_ + grow] = val;
                }
            }
}

// ---------------------------------------------------------------- flash attention
// qkv planes (B,9,K,64) bf16: Q=h, K=3+h, V=6+h. Block = 128 q rows, wave = 16 rows.
__global__ __launch_bounds__(256) void attn_kernel(
    const __bf16* __restrict__ qkv, __bf16* __restrict__ attn_out) {
    __shared__ __bf16 Kl[32][64];
    __shared__ __bf16 Vt[64][32];
    __shared__ __bf16 Pl[8][16][32];
    int tid = threadIdx.x, lane = tid & 31, wave = tid >> 5;
    int half = lane >> 4, mlane = lane & 15;
    int b = blockIdx.z, h = blockIdx.y;
    int q0 = blockIdx.x * 128 + wave * 16;
    const __bf16* Qp = qkv + (size_t)(b * 9 + h) * KSEL * 64;
    const __bf16* Kp = qkv + (size_t)(b * 9 + 3 + h) * KSEL * 64;
    const __bf16* Vp = qkv + (size_t)(b * 9 + 6 + h) * KSEL * 64;

    // preload Q A-fragments (two 16x32 halves of d)
    v16bf aQ[2];
    {
        const __bf16* qr = Qp + (size_t)(q0 + mlane) * 64;
        #pragma unroll
        for (int dh = 0; dh < 2; ++dh) {
            v8bf lo = *(const v8bf*)(qr + dh * 32 + half * 16);
            v8bf hi = *(const v8bf*)(qr + dh * 32 + half * 16 + 8);
            #pragma unroll
            for (int j = 0; j < 8; ++j) { aQ[dh][j] = lo[j]; aQ[dh][8 + j] = hi[j]; }
        }
    }

    float m_i[8], l_i[8];
    v8f o[4];
    #pragma unroll
    for (int r = 0; r < 8; ++r) { m_i[r] = -1e30f; l_i[r] = 0.f; }
    #pragma unroll
    for (int d = 0; d < 4; ++d)
        #pragma unroll
        for (int r = 0; r < 8; ++r) o[d][r] = 0.f;

    const float scale = 0.125f;   // HD^-0.5

    for (int kt = 0; kt < KSEL; kt += 32) {
        // stage K (32x64) via async global->LDS; V^T (64x32) via VGPR transpose
        {
            int r = tid >> 3, cc = (tid & 7) * 8;   // 256 slots of 8 elems
            async_g2l_16B(Kp, ((kt + r) * 64 + cc) * 2, &Kl[r][cc]);
            #pragma unroll
            for (int i = 0; i < 8; ++i) {
                int e = tid * 8 + i;
                int kk = e >> 6, d = e & 63;
                Vt[d][kk] = Vp[(size_t)(kt + kk) * 64 + d];
            }
        }
        wait_async0();
        __syncthreads();

        // S = Q K^T (16 x 32)
        v8f s[2];
        #pragma unroll
        for (int ns = 0; ns < 2; ++ns)
            #pragma unroll
            for (int r = 0; r < 8; ++r) s[ns][r] = 0.f;
        #pragma unroll
        for (int ns = 0; ns < 2; ++ns)
            #pragma unroll
            for (int dh = 0; dh < 2; ++dh) {
                v16bf bk = *(const v16bf*)&Kl[ns * 16 + mlane][dh * 32 + half * 16];
                s[ns] = __builtin_amdgcn_wmma_f32_16x16x32_bf16(
                    false, aQ[dh], false, bk, (short)0, s[ns], false, false);
            }

        // online softmax (DPP 16-lane reductions), write P into wave-private LDS
        #pragma unroll
        for (int r = 0; r < 8; ++r) {
            float s0 = s[0][r] * scale, s1 = s[1][r] * scale;
            float mx = red16_max(fmaxf(s0, s1));
            float mn = fmaxf(m_i[r], mx);
            float p0 = __expf(s0 - mn), p1 = __expf(s1 - mn);
            float rs = red16_sum(p0 + p1);
            float alpha = __expf(m_i[r] - mn);
            l_i[r] = l_i[r] * alpha + rs;
            m_i[r] = mn;
            #pragma unroll
            for (int d = 0; d < 4; ++d) o[d][r] *= alpha;
            int mrow = r + half * 8;
            Pl[wave][mrow][mlane]      = (__bf16)p0;
            Pl[wave][mrow][16 + mlane] = (__bf16)p1;
        }

        // O += P V (wave-private P; compiler inserts ds waits)
        v16bf aP;
        {
            const __bf16* base = &Pl[wave][mlane][0];
            v8bf lo = *(const v8bf*)(base + half * 16);
            v8bf hi = *(const v8bf*)(base + half * 16 + 8);
            #pragma unroll
            for (int j = 0; j < 8; ++j) { aP[j] = lo[j]; aP[8 + j] = hi[j]; }
        }
        #pragma unroll
        for (int d = 0; d < 4; ++d) {
            v16bf bv = *(const v16bf*)&Vt[d * 16 + mlane][half * 16];
            o[d] = __builtin_amdgcn_wmma_f32_16x16x32_bf16(
                false, aP, false, bv, (short)0, o[d], false, false);
        }
        __syncthreads();
    }

    #pragma unroll
    for (int d = 0; d < 4; ++d)
        #pragma unroll
        for (int r = 0; r < 8; ++r) {
            int grow = q0 + r + half * 8;
            attn_out[((size_t)b * KSEL + grow) * C_ + h * HD + d * 16 + mlane] =
                (__bf16)(o[d][r] / l_i[r]);
        }
}

// ---------------------------------------------------------------- launcher
extern "C" void kernel_launch(void* const* d_in, const int* in_sizes, int n_in,
                              void* d_out, int out_size, void* d_ws, size_t ws_size,
                              hipStream_t stream) {
    const float* x      = (const float*)d_in[0];
    const int*   indices= (const int*)d_in[1];
    const float* ln1_w  = (const float*)d_in[2];
    const float* ln1_b  = (const float*)d_in[3];
    const float* qkv_w  = (const float*)d_in[4];
    const float* proj_w = (const float*)d_in[5];
    const float* proj_b = (const float*)d_in[6];
    const float* ln2_w  = (const float*)d_in[7];
    const float* ln2_b  = (const float*)d_in[8];
    const float* fc1_w  = (const float*)d_in[9];
    const float* fc1_b  = (const float*)d_in[10];
    const float* dw_w   = (const float*)d_in[11];
    const float* fc2_w  = (const float*)d_in[12];
    const float* fc2_b  = (const float*)d_in[13];
    float* out = (float*)d_out;

    // workspace carve-up (all offsets 256B aligned)
    char* p = (char*)d_ws;
    auto take = [&](size_t bytes) { char* r = p; p += (bytes + 255) & ~(size_t)255; return r; };
    float*  y     = (float*) take((size_t)B_ * N_ * C_ * 4);       // LN1 out / scatter target
    float*  t2f   = (float*) take((size_t)B_ * N_ * C_ * 4);       // post-LN2 fp32
    __bf16* t2b   = (__bf16*)take((size_t)B_ * N_ * C_ * 2);
    __bf16* sel   = (__bf16*)take((size_t)B_ * KSEL * C_ * 2);
    __bf16* qkvb  = (__bf16*)take((size_t)B_ * 9 * KSEL * 64 * 2);
    __bf16* attnb = (__bf16*)take((size_t)B_ * KSEL * C_ * 2);
    __bf16* h1    = (__bf16*)take((size_t)B_ * HID_ * N_ * 2);
    __bf16* h1g   = (__bf16*)take((size_t)B_ * N_ * HID_ * 2);
    __bf16* wq    = (__bf16*)take((size_t)3 * C_ * C_ * 2);
    __bf16* wp    = (__bf16*)take((size_t)C_ * C_ * 2);
    __bf16* w1    = (__bf16*)take((size_t)HID_ * C_ * 2);
    __bf16* w2    = (__bf16*)take((size_t)C_ * HID_ * 2);

    // weights -> bf16
    f32_to_bf16_kernel<<<(3 * C_ * C_ + 255) / 256, 256, 0, stream>>>(qkv_w, wq, 3 * C_ * C_);
    f32_to_bf16_kernel<<<(C_ * C_ + 255) / 256, 256, 0, stream>>>(proj_w, wp, C_ * C_);
    f32_to_bf16_kernel<<<(HID_ * C_ + 255) / 256, 256, 0, stream>>>(fc1_w, w1, HID_ * C_);
    f32_to_bf16_kernel<<<(C_ * HID_ + 255) / 256, 256, 0, stream>>>(fc2_w, w2, C_ * HID_);

    // LN1
    ln1_kernel<<<(B_ * N_ * 32) / 256, 256, 0, stream>>>(x, ln1_w, ln1_b, y);
    // gather
    gather_kernel<<<B_ * KSEL, C_, 0, stream>>>(y, indices, sel);
    // QKV GEMM: (2048x192)@(192x576)
    gemm_kernel<0><<<dim3(KSEL / 128, 576 / 64, B_), 256, 0, stream>>>(
        sel, wq, nullptr, KSEL, C_, qkvb, nullptr, nullptr);
    // attention
    attn_kernel<<<dim3(KSEL / 128, HEADS, B_), 256, 0, stream>>>(qkvb, attnb);
    // proj GEMM + scatter into y
    gemm_kernel<1><<<dim3(KSEL / 128, C_ / 64, B_), 256, 0, stream>>>(
        attnb, wp, proj_b, KSEL, C_, y, indices, nullptr);
    // residual + LN2
    resid_ln2_kernel<<<(B_ * N_ * 32) / 256, 256, 0, stream>>>(x, y, ln2_w, ln2_b, t2f, t2b);
    // fc1 GEMM: (16384x192)@(192x768) -> transposed bf16 planes
    gemm_kernel<2><<<dim3(N_ / 128, HID_ / 64, B_), 256, 0, stream>>>(
        t2b, w1, fc1_b, N_, C_, h1, nullptr, nullptr);
    // depthwise conv + GELU
    dwconv_gelu_kernel<<<(int)(((size_t)B_ * HID_ * N_) / 256), 256, 0, stream>>>(h1, dw_w, h1g);
    // fc2 GEMM + bias + residual -> final (B,C,H,W)
    gemm_kernel<3><<<dim3(N_ / 128, C_ / 64, B_), 256, 0, stream>>>(
        h1g, w2, fc2_b, N_, HID_, out, nullptr, t2f);
}
